// ShiftedWindowAttentionInf_44212393345679
// MI455X (gfx1250) — matrix operand order repaired and checked by reading, not compiled, not run on hardware
//
#include <hip/hip_runtime.h>

typedef __attribute__((ext_vector_type(16))) _Float16 v16h;
typedef __attribute__((ext_vector_type(8)))  _Float16 v8h;
typedef __attribute__((ext_vector_type(8)))  float    v8f;

#define WIN   7
#define SHIFT 3
#define NH    4
#define HD    32
#define CDIM  128
#define HH    56
#define WW2   56
#define NTOK  49
#define WB    2048            // B(32) * nW(64)
#define ROWS  (WB * NTOK)     // 100352, divisible by 16
#define MT    (ROWS / 16)     // 6272

// ---- WMMA fragment loaders (per ISA 16-bit layouts, wave32) ----
// A: 16x32 (MxK). lane&15 = row; lanes>=16 take K+8; vgprs 4..7 take K+16.
static __device__ __forceinline__ v16h load_afrag(const _Float16* tile, int ld, int lane) {
  const _Float16* p = tile + (lane & 15) * ld + ((lane >> 4) << 3);
  v8h lo = *(const v8h*)(p);
  v8h hi = *(const v8h*)(p + 16);
  return __builtin_shufflevector(lo, hi, 0,1,2,3,4,5,6,7,8,9,10,11,12,13,14,15);
}
// B: 32x16 (KxN) read from transposed source (rows indexed by N, cols by K).
// lane&15 = column n; lanes>=16 take K+8; elements 8..15 take K+16.
static __device__ __forceinline__ v16h load_bfrag(const _Float16* tileT, int ld, int lane) {
  const _Float16* p = tileT + (lane & 15) * ld + ((lane >> 4) << 3);
  v8h lo = *(const v8h*)(p);
  v8h hi = *(const v8h*)(p + 16);
  return __builtin_shufflevector(lo, hi, 0,1,2,3,4,5,6,7,8,9,10,11,12,13,14,15);
}

// ---- Kernel 0: fp32 -> f16 weight conversion ----
__global__ void cvt_weights(const float* __restrict__ qkv_w, const float* __restrict__ proj_w,
                            _Float16* __restrict__ qwf, _Float16* __restrict__ pwf) {
  int i = blockIdx.x * 256 + threadIdx.x;
  if (i < 384 * CDIM) qwf[i] = (_Float16)qkv_w[i];
  int j = i - 384 * CDIM;
  if (j >= 0 && j < CDIM * CDIM) pwf[j] = (_Float16)proj_w[j];
}

// ---- Kernel 1: roll(-3,-3) + window partition + f16 convert ----
__global__ void gather_x(const float* __restrict__ x, _Float16* __restrict__ xw) {
  int r = blockIdx.x;                 // 0..ROWS-1
  int c = threadIdx.x;                // 0..127
  int wb = r / NTOK; int t = r - wb * NTOK;
  int b = wb >> 6; int wid = wb & 63;
  int wh = wid >> 3, ww = wid & 7;
  int th = t / WIN, tw = t - th * WIN;
  int h = wh * WIN + th + SHIFT; if (h >= HH)  h -= HH;
  int w = ww * WIN + tw + SHIFT; if (w >= WW2) w -= WW2;
  xw[(size_t)r * CDIM + c] = (_Float16)x[(((size_t)b * HH + h) * WW2 + w) * CDIM + c];
}

// ---- Kernel 2: fused QKV projection. Wave tile = 16x64 (A-frag reused x4) ----
__global__ void __launch_bounds__(128) qkv_gemm(const _Float16* __restrict__ xw,
                                                const _Float16* __restrict__ qwf,
                                                const float* __restrict__ qkv_b,
                                                _Float16* __restrict__ qbuf,
                                                _Float16* __restrict__ kbuf,
                                                _Float16* __restrict__ vbuf) {
  int lane = threadIdx.x & 31;
  int wave = threadIdx.x >> 5;
  int m0 = (blockIdx.x * 4 + wave) * 16;
  int o0 = blockIdx.y * 64;           // gridDim.y = 6
  v8f cc[4] = {{}, {}, {}, {}};
#pragma unroll
  for (int k0 = 0; k0 < CDIM; k0 += 32) {
    v16h a = load_afrag(xw + (size_t)m0 * CDIM + k0, CDIM, lane);
#pragma unroll
    for (int nt = 0; nt < 4; nt++) {
      v16h b = load_bfrag(qwf + (size_t)(o0 + nt * 16) * CDIM + k0, CDIM, lane);
      cc[nt] = __builtin_amdgcn_wmma_f32_16x16x32_f16(false, a, false, b, (short)0, cc[nt], false, false);
    }
  }
  int n = lane & 15;
  int rb = (lane >> 4) << 3;
  // hoist row -> (wb, t) mapping out of the nt loop
  int base[8];
#pragma unroll
  for (int r = 0; r < 8; r++) {
    int row = m0 + rb + r;
    int wb = row / NTOK; int t = row - wb * NTOK;
    base[r] = wb * (NH * NTOK * HD) + t * HD;   // + head*NTOK*HD + dim
  }
#pragma unroll
  for (int nt = 0; nt < 4; nt++) {
    int o = o0 + nt * 16 + n;
    float bias = qkv_b[o];
    int which = o >> 7;               // constant per 64-wide block
    int head  = (o & 127) >> 5;
    int dim   = o & 31;
    float scale = (which == 0) ? 0.17677669529663687f : 1.0f; // 1/sqrt(32) folded into q
    _Float16* dst = (which == 0) ? qbuf : (which == 1) ? kbuf : vbuf;
    int hoff = head * (NTOK * HD) + dim;
#pragma unroll
    for (int r = 0; r < 8; r++)
      dst[(size_t)base[r] + hoff] = (_Float16)((cc[nt][r] + bias) * scale);
  }
}

// ---- Kernel 3: per-(window,head) attention: S=QK^T, mask+bias+softmax, P@V ----
__global__ void __launch_bounds__(32) attn_kernel(const _Float16* __restrict__ qbuf,
                                                  const _Float16* __restrict__ kbuf,
                                                  const _Float16* __restrict__ vbuf,
                                                  const float* __restrict__ rel_bias,
                                                  _Float16* __restrict__ attn_out) {
  // Q/K staging is dead once S is built; P reuses its space (saves 8KB LDS -> occupancy)
  __shared__ __align__(32) union {
    struct { _Float16 qs[64 * HD]; _Float16 ks[64 * HD]; } qk;
    _Float16 P[64 * 64];
  } u;
  __shared__ __align__(32) _Float16 vt[HD * 64];  // V transposed: [dim][token]
  __shared__ float S[64 * 64];
  __shared__ int regS[64];

  int lane = threadIdx.x;
  int bh = blockIdx.x;
  int wb = bh >> 2, head = bh & 3;
  const _Float16* qg = qbuf + (((size_t)wb * NH + head) * NTOK) * HD;
  const _Float16* kg = kbuf + (((size_t)wb * NH + head) * NTOK) * HD;
  const _Float16* vg = vbuf + (((size_t)wb * NH + head) * NTOK) * HD;
  int wid = wb & 63; int wh = wid >> 3, ww = wid & 7;

  for (int t = lane; t < 64; t += 32) {
    if (t < NTOK) {
      int th = t / WIN, tw = t - th * WIN;
      int hs = wh * WIN + th, ws = ww * WIN + tw;
      int ch = (hs >= HH - WIN) + (hs >= HH - SHIFT);
      int cw = (ws >= WW2 - WIN) + (ws >= WW2 - SHIFT);
      regS[t] = ch * 3 + cw;
    } else regS[t] = -1;
#pragma unroll
    for (int j = 0; j < 4; j++) {
      v8h zq = {}, zk = {}, zv = {};
      if (t < NTOK) {
        zq = *(const v8h*)(qg + t * HD + j * 8);
        zk = *(const v8h*)(kg + t * HD + j * 8);
        zv = *(const v8h*)(vg + t * HD + j * 8);
      }
      *(v8h*)(u.qk.qs + t * HD + j * 8) = zq;
      *(v8h*)(u.qk.ks + t * HD + j * 8) = zk;
#pragma unroll
      for (int e = 0; e < 8; e++) vt[(j * 8 + e) * 64 + t] = zv[e];
    }
  }
  __syncthreads();

  // S = Q (64x32) * K^T -> 64x64, single K-step WMMA per tile
#pragma unroll
  for (int mt = 0; mt < 4; mt++) {
    v16h a = load_afrag(u.qk.qs + mt * 16 * HD, HD, lane);
#pragma unroll
    for (int nt = 0; nt < 4; nt++) {
      v16h b = load_bfrag(u.qk.ks + nt * 16 * HD, HD, lane);
      v8f c = __builtin_amdgcn_wmma_f32_16x16x32_f16(false, a, false, b, (short)0, (v8f){}, false, false);
      int n = lane & 15, rb = (lane >> 4) << 3;
#pragma unroll
      for (int r = 0; r < 8; r++) S[(mt * 16 + rb + r) * 64 + nt * 16 + n] = c[r];
    }
  }
  __syncthreads();

  // bias + shift-mask + softmax; P (aliasing qs/ks) holds normalized f16 probs
  const float* bg = rel_bias + (size_t)head * NTOK * NTOK;
  for (int row = lane; row < 64; row += 32) {
    if (row < NTOK) {
      int ri = regS[row];
      float mx = -1e30f;
      for (int j = 0; j < NTOK; j++) {
        float v = S[row * 64 + j] + bg[row * NTOK + j] + ((regS[j] != ri) ? -100.0f : 0.0f);
        S[row * 64 + j] = v;
        mx = fmaxf(mx, v);
      }
      float sum = 0.f;
      for (int j = 0; j < NTOK; j++) {
        float e = __expf(S[row * 64 + j] - mx);
        S[row * 64 + j] = e;
        sum += e;
      }
      float rinv = 1.0f / sum;
      for (int j = 0; j < 64; j++)
        u.P[row * 64 + j] = (j < NTOK) ? (_Float16)(S[row * 64 + j] * rinv) : (_Float16)0.f;
    } else {
      for (int j = 0; j < 64; j++) u.P[row * 64 + j] = (_Float16)0.f;
    }
  }
  __syncthreads();

  // out = P (64x64) @ V (64x32); B-frags come contiguously from vt
#pragma unroll
  for (int mt = 0; mt < 4; mt++) {
#pragma unroll
    for (int nt = 0; nt < 2; nt++) {
      v8f c = {};
#pragma unroll
      for (int k0 = 0; k0 < 64; k0 += 32) {
        v16h a = load_afrag(u.P + mt * 16 * 64 + k0, 64, lane);
        v16h b = load_bfrag(vt + nt * 16 * 64 + k0, 64, lane);
        c = __builtin_amdgcn_wmma_f32_16x16x32_f16(false, a, false, b, (short)0, c, false, false);
      }
      int n = lane & 15, rb = (lane >> 4) << 3;
#pragma unroll
      for (int r = 0; r < 8; r++) {
        int t = mt * 16 + rb + r;
        if (t < NTOK)
          attn_out[((size_t)wb * NTOK + t) * CDIM + head * HD + nt * 16 + n] = (_Float16)c[r];
      }
    }
  }
}

// ---- Kernel 4: output projection. Wave tile = 16x64; fused bias+merge+roll ----
__global__ void __launch_bounds__(128) proj_gemm(const _Float16* __restrict__ ain,
                                                 const _Float16* __restrict__ pwf,
                                                 const float* __restrict__ proj_b,
                                                 float* __restrict__ out) {
  int lane = threadIdx.x & 31;
  int wave = threadIdx.x >> 5;
  int m0 = (blockIdx.x * 4 + wave) * 16;
  int o0 = blockIdx.y * 64;           // gridDim.y = 2
  v8f cc[4] = {{}, {}, {}, {}};
#pragma unroll
  for (int k0 = 0; k0 < CDIM; k0 += 32) {
    v16h a = load_afrag(ain + (size_t)m0 * CDIM + k0, CDIM, lane);
#pragma unroll
    for (int nt = 0; nt < 4; nt++) {
      v16h b = load_bfrag(pwf + (size_t)(o0 + nt * 16) * CDIM + k0, CDIM, lane);
      cc[nt] = __builtin_amdgcn_wmma_f32_16x16x32_f16(false, a, false, b, (short)0, cc[nt], false, false);
    }
  }
  int n = lane & 15;
  int rb = (lane >> 4) << 3;
  // hoist row -> output pixel mapping out of the nt loop
  size_t obase[8];
#pragma unroll
  for (int r = 0; r < 8; r++) {
    int row = m0 + rb + r;
    int wb = row / NTOK; int t = row - wb * NTOK;
    int b = wb >> 6; int wid = wb & 63;
    int wh = wid >> 3, ww = wid & 7;
    int th = t / WIN, tw = t - th * WIN;
    int h = wh * WIN + th + SHIFT; if (h >= HH)  h -= HH;
    int w = ww * WIN + tw + SHIFT; if (w >= WW2) w -= WW2;
    obase[r] = (((size_t)b * HH + h) * WW2 + w) * CDIM;
  }
#pragma unroll
  for (int nt = 0; nt < 4; nt++) {
    int cc_col = o0 + nt * 16 + n;
    float bias = proj_b[cc_col];
#pragma unroll
    for (int r = 0; r < 8; r++)
      out[obase[r] + cc_col] = cc[nt][r] + bias;
  }
}

extern "C" void kernel_launch(void* const* d_in, const int* in_sizes, int n_in,
                              void* d_out, int out_size, void* d_ws, size_t ws_size,
                              hipStream_t stream) {
  (void)in_sizes; (void)n_in; (void)out_size; (void)ws_size;
  const float* x      = (const float*)d_in[0];
  const float* qkv_w  = (const float*)d_in[1];
  const float* qkv_b  = (const float*)d_in[2];
  const float* proj_w = (const float*)d_in[3];
  const float* proj_b = (const float*)d_in[4];
  const float* relb   = (const float*)d_in[5];
  float* out = (float*)d_out;

  char* ws = (char*)d_ws;
  const size_t SEG = (size_t)ROWS * CDIM * sizeof(_Float16);   // 25,690,112 B
  _Float16* xw   = (_Float16*)(ws);                // gathered x; later reused as attn output
  _Float16* qbuf = (_Float16*)(ws + SEG);
  _Float16* kbuf = (_Float16*)(ws + 2 * SEG);
  _Float16* vbuf = (_Float16*)(ws + 3 * SEG);
  _Float16* qwf  = (_Float16*)(ws + 4 * SEG);
  _Float16* pwf  = (_Float16*)(ws + 4 * SEG + (size_t)384 * CDIM * sizeof(_Float16));

  cvt_weights<<<256, 256, 0, stream>>>(qkv_w, proj_w, qwf, pwf);
  gather_x<<<ROWS, CDIM, 0, stream>>>(x, xw);
  qkv_gemm<<<dim3(MT / 4, 6), 128, 0, stream>>>(xw, qwf, qkv_b, qbuf, kbuf, vbuf);
  attn_kernel<<<WB * NH, 32, 0, stream>>>(qbuf, kbuf, vbuf, relb, xw);
  proj_gemm<<<dim3(MT / 4, 2), 128, 0, stream>>>(xw, pwf, proj_b, out);
}
